// LSMPool_67199058313451
// MI455X (gfx1250) — compile-verified, matrix operand order checked by reading, not run on hardware
//
#include <hip/hip_runtime.h>
#include <hip/hip_bf16.h>

// ---------------------------------------------------------------------------
// LSM network on MI455X (gfx1250, wave32, WMMA).
//
//   1. Split W1/W2 into bf16 hi/lo pairs (spikes are exactly bf16, so
//      spk@W = spk@Whi + spk@Wlo recovers ~fp32-class accuracy on the f32
//      WMMA accumulator).  Convert x (binary) to bf16.
//   2. Stage 1: 64 sequential GEMM+LIF kernels.  Per step t the A-matrix is
//      [x_t | spk_{t-1}] (two K segments, 1024 + 4096), B = W1hi/W1lo.
//      LIF fused in epilogue; spikes stored as bf16 straight into spk_rec[t].
//   3. Stage 2: ONE big GEMM (T*B=16384 x 1024 x K=4096) of spk_rec @ W2^T
//      into cur2 (f32), then an elementwise LIF scan over T.
//
// Tiling (this round): WG = 128 threads = 4 waves in a 2x2 grid over a
// 128x128 tile; each wave owns a 64x64 tile = 4x4 f32 accumulators.
// Per K-step per wave: 32 WMMAs vs 24 ds_load_b128  -> WMMA-bound.
// ---------------------------------------------------------------------------

typedef __attribute__((ext_vector_type(16))) __bf16 v16bf;
typedef __attribute__((ext_vector_type(8)))  float  v8f;

#define TM   128    // workgroup M tile
#define TN   128    // workgroup N tile
#define TK   32     // K step (WMMA bf16 K)
#define LSTR 48     // LDS row stride in shorts (96B: 16/32B aligned, bank-spread)

__device__ __forceinline__ unsigned short f2bf(float f) {
    unsigned int u = __float_as_uint(f);
    unsigned int r = u + 0x7FFFu + ((u >> 16) & 1u);   // round-to-nearest-even
    return (unsigned short)(r >> 16);
}
__device__ __forceinline__ float bf2f(unsigned short h) {
    return __uint_as_float(((unsigned int)h) << 16);
}

// ---------------------------------------------------------------------------
// Setup kernels
// ---------------------------------------------------------------------------
__global__ void split_hilo_kernel(const float* __restrict__ w,
                                  unsigned short* __restrict__ hi,
                                  unsigned short* __restrict__ lo, int n) {
    int i = blockIdx.x * 256 + threadIdx.x;
    if (i < n) {
        float f = w[i];
        unsigned short h = f2bf(f);
        hi[i] = h;
        lo[i] = f2bf(f - bf2f(h));
    }
}

__global__ void cvt_bf16_kernel(const float* __restrict__ x,
                                unsigned short* __restrict__ y, int n) {
    int i = blockIdx.x * 256 + threadIdx.x;
    if (i < n) y[i] = f2bf(x[i]);            // x is binary {0,1}: exact
}

__global__ void zero_f32_kernel(float* __restrict__ p, int n) {
    int i = blockIdx.x * 256 + threadIdx.x;
    if (i < n) p[i] = 0.0f;
}
__global__ void zero_u16_kernel(unsigned short* __restrict__ p, int n) {
    int i = blockIdx.x * 256 + threadIdx.x;
    if (i < n) p[i] = 0;
}

// ---------------------------------------------------------------------------
// GEMM with split-bf16 weights.  C[M,N] = A(bf16 spikes) @ (Bhi+Blo) layout.
//   A given as two K-segments (input spikes + recurrent spikes).
//   mode 0: fused LIF epilogue -> mem update + bf16 spike store
//   mode 1: plain f32 store (+bias)
// ---------------------------------------------------------------------------
__global__ __launch_bounds__(128)
void gemm_splitw_kernel(const unsigned short* __restrict__ A0, int lenK0, int lda0,
                        const unsigned short* __restrict__ A1, int lenK1, int lda1,
                        const unsigned short* __restrict__ Bhi,
                        const unsigned short* __restrict__ Blo, int ldb,
                        int N,
                        float* __restrict__ mem,            // mode 0
                        unsigned short* __restrict__ spkOut,// mode 0 (bf16)
                        float* __restrict__ curOut,         // mode 1
                        const float* __restrict__ bias,     // mode 1
                        int mode) {
    __shared__ unsigned short As[TM * LSTR];
    __shared__ unsigned short Bh[TN * LSTR];
    __shared__ unsigned short Bl[TN * LSTR];

    const int tid   = threadIdx.x;
    const int lane  = tid & 31;
    const int waveM = (tid >> 5) >> 1;   // 0..1
    const int waveN = (tid >> 5) & 1;    // 0..1
    const int mBlock = blockIdx.y * TM;
    const int nBlock = blockIdx.x * TN;

    const int lrow = lane & 15;          // fragment row within 16
    const int koff = (lane >> 4) << 4;   // K offset 0/16 for lane halves

    v8f acc[4][4];
    const v8f vzero = {0.f, 0.f, 0.f, 0.f, 0.f, 0.f, 0.f, 0.f};
#pragma unroll
    for (int im = 0; im < 4; ++im)
#pragma unroll
        for (int in = 0; in < 4; ++in) acc[im][in] = vzero;

    for (int s = 0; s < 2; ++s) {
        const unsigned short* Ap = s ? A1 : A0;
        const int lda   = s ? lda1 : lda0;
        const int lenK  = s ? lenK1 : lenK0;
        const int kOffB = s ? lenK0 : 0;

        for (int k = 0; k < lenK; k += TK) {
            __syncthreads();
            // ---- stage tiles: each 128 x 32 bf16 = 4 passes of 128x8 elems ----
#pragma unroll
            for (int p = 0; p < 4; ++p) {
                int e = p * 1024 + tid * 8;
                int row = e >> 5, col = e & 31;
                *reinterpret_cast<uint4*>(As + row * LSTR + col) =
                    *reinterpret_cast<const uint4*>(
                        Ap + (size_t)(mBlock + row) * lda + k + col);
                size_t g = (size_t)(nBlock + row) * ldb + kOffB + k + col;
                *reinterpret_cast<uint4*>(Bh + row * LSTR + col) =
                    *reinterpret_cast<const uint4*>(Bhi + g);
                *reinterpret_cast<uint4*>(Bl + row * LSTR + col) =
                    *reinterpret_cast<const uint4*>(Blo + g);
            }
            __syncthreads();

            // ---- compute: 32 v_wmma_f32_16x16x32_bf16 per wave per K step ----
            v16bf af[4];
#pragma unroll
            for (int im = 0; im < 4; ++im)
                af[im] = *reinterpret_cast<const v16bf*>(
                    As + (waveM * 64 + im * 16 + lrow) * LSTR + koff);
#pragma unroll
            for (int in = 0; in < 4; ++in) {
                const int boff = (waveN * 64 + in * 16 + lrow) * LSTR + koff;
                v16bf bh = *reinterpret_cast<const v16bf*>(Bh + boff);
                v16bf bl = *reinterpret_cast<const v16bf*>(Bl + boff);
#pragma unroll
                for (int im = 0; im < 4; ++im) {
                    acc[im][in] = __builtin_amdgcn_wmma_f32_16x16x32_bf16(
                        false, af[im], false, bh, (short)0, acc[im][in], false, false);
                    acc[im][in] = __builtin_amdgcn_wmma_f32_16x16x32_bf16(
                        false, af[im], false, bl, (short)0, acc[im][in], false, false);
                }
            }
        }
    }

    // ---- epilogue ----
    // C 16x16 f32 layout: VGPR r, lanes 0-15 -> M=r, lanes 16-31 -> M=r+8; N=lane%16.
    const int rBase = mBlock + waveM * 64 + ((lane >> 4) << 3);
    const int cBase = nBlock + waveN * 64 + (lane & 15);
#pragma unroll
    for (int im = 0; im < 4; ++im) {
#pragma unroll
        for (int in = 0; in < 4; ++in) {
#pragma unroll
            for (int r = 0; r < 8; ++r) {
                int row = rBase + im * 16 + r;
                int col = cBase + in * 16;
                size_t idx = (size_t)row * N + col;
                float v = acc[im][in][r];
                if (mode == 0) {
                    float m = mem[idx] + v;          // beta=1 integrate
                    m = m > 0.f ? m : 0.f;           // state_quant = relu
                    float spk = m > 1.0f ? 1.f : 0.f;
                    mem[idx] = m - spk;              // reset-by-subtraction
                    spkOut[idx] = spk != 0.f ? (unsigned short)0x3F80
                                             : (unsigned short)0;
                } else {
                    curOut[idx] = v + bias[col];
                }
            }
        }
    }
}

// ---------------------------------------------------------------------------
// Stage-2 LIF scan over T (per (b,out) neuron, sequential over time).
// ---------------------------------------------------------------------------
__global__ void readout_scan_kernel(const float* __restrict__ cur2,
                                    float* __restrict__ out, int BO, int T) {
    int i = blockIdx.x * 256 + threadIdx.x;
    if (i >= BO) return;
    float mem = 0.f;
    for (int t = 0; t < T; ++t) {
        float m = mem + cur2[(size_t)t * BO + i];
        m = m > 0.f ? m : 0.f;
        float s = m > 1.f ? 1.f : 0.f;
        out[(size_t)t * BO + i] = s;
        mem = m - s;
    }
}

// ---------------------------------------------------------------------------
extern "C" void kernel_launch(void* const* d_in, const int* in_sizes, int n_in,
                              void* d_out, int out_size, void* d_ws, size_t ws_size,
                              hipStream_t stream) {
    (void)in_sizes; (void)n_in; (void)out_size; (void)ws_size;
    constexpr int T = 64, B = 256, IN = 1024, HID = 4096, OUT = 1024;
    constexpr int K1 = IN + HID;   // 5120

    const float* x  = (const float*)d_in[0];
    const float* W1 = (const float*)d_in[1];
    const float* W2 = (const float*)d_in[2];
    const float* b2 = (const float*)d_in[3];
    float* out = (float*)d_out;

    // workspace carve-out (~342 MB)
    char* ws = (char*)d_ws;
    size_t off = 0;
    auto carve = [&](size_t bytes) {
        void* p = ws + off;
        off += (bytes + 255) & ~(size_t)255;
        return p;
    };
    unsigned short* W1hi   = (unsigned short*)carve((size_t)HID * K1 * 2);
    unsigned short* W1lo   = (unsigned short*)carve((size_t)HID * K1 * 2);
    unsigned short* W2hi   = (unsigned short*)carve((size_t)OUT * HID * 2);
    unsigned short* W2lo   = (unsigned short*)carve((size_t)OUT * HID * 2);
    unsigned short* xbf    = (unsigned short*)carve((size_t)T * B * IN * 2);
    unsigned short* spk0   = (unsigned short*)carve((size_t)B * HID * 2);
    unsigned short* spkRec = (unsigned short*)carve((size_t)T * B * HID * 2);
    float*          mem1   = (float*)carve((size_t)B * HID * 4);
    float*          cur2   = (float*)carve((size_t)T * B * OUT * 4);

    // ---- setup ----
    {
        int n1 = HID * K1;
        split_hilo_kernel<<<(n1 + 255) / 256, 256, 0, stream>>>(W1, W1hi, W1lo, n1);
        int n2 = OUT * HID;
        split_hilo_kernel<<<(n2 + 255) / 256, 256, 0, stream>>>(W2, W2hi, W2lo, n2);
        int nx = T * B * IN;
        cvt_bf16_kernel<<<(nx + 255) / 256, 256, 0, stream>>>(x, xbf, nx);
        int nm = B * HID;
        zero_f32_kernel<<<(nm + 255) / 256, 256, 0, stream>>>(mem1, nm);
        zero_u16_kernel<<<(nm + 255) / 256, 256, 0, stream>>>(spk0, nm);
    }

    // ---- stage 1: recurrent reservoir, sequential over T ----
    {
        dim3 blk(128);
        dim3 grd(HID / TN, B / TM);   // 32 x 2 workgroups
        for (int t = 0; t < T; ++t) {
            const unsigned short* Aprev =
                (t == 0) ? spk0 : (spkRec + (size_t)(t - 1) * B * HID);
            gemm_splitw_kernel<<<grd, blk, 0, stream>>>(
                xbf + (size_t)t * B * IN, IN, IN,   // segment 0: x_t
                Aprev, HID, HID,                    // segment 1: spk_{t-1}
                W1hi, W1lo, K1,
                HID,
                mem1, spkRec + (size_t)t * B * HID, // LIF epilogue
                nullptr, nullptr, /*mode=*/0);
        }
    }

    // ---- stage 2: one big readout GEMM over all timesteps ----
    {
        dim3 blk(128);
        dim3 grd(OUT / TN, (T * B) / TM);   // 8 x 128 workgroups
        gemm_splitw_kernel<<<grd, blk, 0, stream>>>(
            spkRec, HID, HID,
            nullptr, 0, 1,
            W2hi, W2lo, HID,
            OUT,
            nullptr, nullptr,
            cur2, b2, /*mode=*/1);
    }

    // ---- stage 2 LIF scan over T ----
    {
        int bo = B * OUT;
        readout_scan_kernel<<<(bo + 255) / 256, 256, 0, stream>>>(cur2, out, bo, T);
    }
}
